// MMDLoss_3616362463816
// MI455X (gfx1250) — matrix operand order repaired and checked by reading, not compiled, not run on hardware
//
#include <hip/hip_runtime.h>
#include <hip/hip_bf16.h>

typedef __attribute__((ext_vector_type(8)))  __bf16 v8bf;
typedef __attribute__((ext_vector_type(16))) __bf16 v16bf;
typedef __attribute__((ext_vector_type(8)))  float  v8f;

#define DIM    256
#define NHALF  4096
#define MTOT   8192

// ws layout (float offsets)
#define OFF_SQ    0      // 8192 row squared-norms
#define OFF_S     8192   // 256 column sums
#define OFF_SUMSQ 8448   // scalar sum of squared norms
#define OFF_SCALE 8449   // 1/(4*bandwidth)
#define OFF_QUAD  8452   // 4 quadrant kernel sums

__global__ void mmd_zero(float* __restrict__ ws) {
    int t = threadIdx.x;
    for (int i = 8192 + t; i < 8456; i += 256) ws[i] = 0.0f;
}

// 32 blocks x 256 threads: column sums (coalesced) + per-row ||z||^2 + total.
__global__ void mmd_rowstats(const float* __restrict__ X,
                             const float* __restrict__ Y,
                             float* __restrict__ ws) {
    int t = threadIdx.x;
    int b = blockIdx.x;                    // 256 rows per block; halves are block-aligned
    const float* base = (b < 16) ? (X + (size_t)b * 256 * DIM)
                                 : (Y + (size_t)(b - 16) * 256 * DIM);
    // column partial sums: thread t owns column t (coalesced across threads)
    float cs = 0.0f;
    for (int r = 0; r < 256; ++r) cs += base[(size_t)r * DIM + t];
    atomicAdd(&ws[OFF_S + t], cs);
    // row squared norm: thread t owns row t of this block
    float ss = 0.0f;
    for (int c = 0; c < DIM; ++c) { float x = base[(size_t)t * DIM + c]; ss += x * x; }
    ws[OFF_SQ + b * 256 + t] = ss;
    __shared__ float red[256];
    red[t] = ss;
    __syncthreads();
    for (int off = 128; off; off >>= 1) { if (t < off) red[t] += red[t + off]; __syncthreads(); }
    if (t == 0) atomicAdd(&ws[OFF_SUMSQ], red[0]);
}

// sum(d2) = 2*M*sum(sq) - 2*||colsum||^2  -> bandwidth -> scale = 1/(4*bw)
__global__ void mmd_bw(float* __restrict__ ws) {
    int t = threadIdx.x;
    float v = ws[OFF_S + t];
    __shared__ float red[256];
    red[t] = v * v;
    __syncthreads();
    for (int off = 128; off; off >>= 1) { if (t < off) red[t] += red[t + off]; __syncthreads(); }
    if (t == 0) {
        float sumsq  = ws[OFF_SUMSQ];
        float sum_d2 = 2.0f * (float)MTOT * sumsq - 2.0f * red[0];
        float denom  = (float)((double)MTOT * (double)MTOT - (double)MTOT);
        float bw     = sum_d2 / denom;
        ws[OFF_SCALE] = 1.0f / (4.0f * bw);
    }
}

// 128x128 tile of the 8192x8192 kernel matrix per block, WMMA bf16 hi/lo split.
__launch_bounds__(256, 1)
__global__ void mmd_main(const float* __restrict__ X,
                         const float* __restrict__ Y,
                         float* __restrict__ ws) {
    const int bj = blockIdx.x, bi = blockIdx.y;
    if (bj < bi) return;                    // symmetry: upper-triangular blocks only
    const int rowBase = bi * 128, colBase = bj * 128;

    __shared__ __align__(32) __bf16 sAhi[128][64];   // [m][k]
    __shared__ __align__(32) __bf16 sAlo[128][64];
    __shared__ __align__(32) __bf16 sBhi[64][128];   // [k][n]  (B panel K-major)
    __shared__ __align__(32) __bf16 sBlo[64][128];

    const int t    = threadIdx.x;
    const int lane = t & 31, wave = t >> 5;
    const int wm = wave >> 2, wn = wave & 3;         // wave grid 2x4
    const int half = lane >> 4, l15 = lane & 15;

    const float* Apanel = (rowBase < NHALF) ? X + (size_t)rowBase * DIM
                                            : Y + (size_t)(rowBase - NHALF) * DIM;
    const float* Bpanel = (colBase < NHALF) ? X + (size_t)colBase * DIM
                                            : Y + (size_t)(colBase - NHALF) * DIM;

    v8f acc[4][2];
    #pragma unroll
    for (int mt = 0; mt < 4; ++mt)
        #pragma unroll
        for (int nt = 0; nt < 2; ++nt)
            #pragma unroll
            for (int q = 0; q < 8; ++q) acc[mt][nt][q] = 0.0f;

    for (int kc = 0; kc < 4; ++kc) {                 // K chunks of 64
        #pragma unroll
        for (int e = 0; e < 32; ++e) {               // 8192 elems / 256 threads
            int lid = e * 256 + t;
            int r = lid >> 6, k = lid & 63;
            float xa = Apanel[(size_t)r * DIM + kc * 64 + k];
            __bf16 ha = (__bf16)xa;
            sAhi[r][k] = ha;
            sAlo[r][k] = (__bf16)(xa - (float)ha);
            float xb = Bpanel[(size_t)r * DIM + kc * 64 + k];
            __bf16 hb = (__bf16)xb;
            sBhi[k][r] = hb;                         // transpose into K-major
            sBlo[k][r] = (__bf16)(xb - (float)hb);
        }
        __syncthreads();
        #pragma unroll
        for (int ks = 0; ks < 2; ++ks) {             // two 16x16x32 steps
            const int k0 = ks * 32;
            v16bf bhi[2], blo[2];
            #pragma unroll
            for (int nt = 0; nt < 2; ++nt) {
                // B layout: column N=l15; elems 0..7 -> K=half*8+0..7, elems 8..15 -> K=16+half*8+0..7
                #pragma unroll
                for (int s = 0; s < 16; ++s) {
                    const int kk = (s < 8) ? (half * 8 + s) : (16 + half * 8 + (s - 8));
                    bhi[nt][s] = sBhi[k0 + kk][wn * 32 + nt * 16 + l15];
                    blo[nt][s] = sBlo[k0 + kk][wn * 32 + nt * 16 + l15];
                }
            }
            #pragma unroll
            for (int mt = 0; mt < 4; ++mt) {
                const int m = wm * 64 + mt * 16 + l15;
                // A layout: elems 0..7 -> K=half*8+0..7, elems 8..15 -> K=16+half*8+0..7
                const __bf16* pa = &sAhi[m][k0 + half * 8];
                v8bf a0 = *(const v8bf*)pa;
                v8bf a1 = *(const v8bf*)(pa + 16);
                v16bf ahi = __builtin_shufflevector(a0, a1,
                    0,1,2,3,4,5,6,7,8,9,10,11,12,13,14,15);
                const __bf16* pl = &sAlo[m][k0 + half * 8];
                v8bf c0 = *(const v8bf*)pl;
                v8bf c1 = *(const v8bf*)(pl + 16);
                v16bf alo = __builtin_shufflevector(c0, c1,
                    0,1,2,3,4,5,6,7,8,9,10,11,12,13,14,15);
                #pragma unroll
                for (int nt = 0; nt < 2; ++nt) {
                    acc[mt][nt] = __builtin_amdgcn_wmma_f32_16x16x32_bf16(
                        false, ahi, false, bhi[nt], (short)0, acc[mt][nt], false, false);
                    acc[mt][nt] = __builtin_amdgcn_wmma_f32_16x16x32_bf16(
                        false, ahi, false, blo[nt], (short)0, acc[mt][nt], false, false);
                    acc[mt][nt] = __builtin_amdgcn_wmma_f32_16x16x32_bf16(
                        false, alo, false, bhi[nt], (short)0, acc[mt][nt], false, false);
                }
            }
        }
        __syncthreads();
    }

    // epilogue: d2 -> K = e + e^2 + e^4 + e^8 + e^16, e = exp(-d2/(4bw))
    const float scale = ws[OFF_SCALE];
    const float* sq = ws + OFF_SQ;
    float ksum = 0.0f;
    #pragma unroll
    for (int nt = 0; nt < 2; ++nt) {
        const int j = colBase + wn * 32 + nt * 16 + l15;
        const float sqj = sq[j];
        #pragma unroll
        for (int mt = 0; mt < 4; ++mt) {
            #pragma unroll
            for (int r = 0; r < 8; ++r) {
                const int i = rowBase + wm * 64 + mt * 16 + r + 8 * half;
                float d2 = sq[i] + sqj - 2.0f * acc[mt][nt][r];
                d2 = fmaxf(d2, 0.0f);
                float e1  = __expf(-d2 * scale);
                float e2  = e1 * e1;
                float e4  = e2 * e2;
                float e8  = e4 * e4;
                float e16 = e8 * e8;
                ksum += e1 + e2 + e4 + e8 + e16;
            }
        }
    }

    float* red = (float*)&sAhi[0][0];   // alias A tile (done with it)
    red[t] = ksum;
    __syncthreads();
    for (int off = 128; off; off >>= 1) { if (t < off) red[t] += red[t + off]; __syncthreads(); }
    if (t == 0) {
        float s = red[0];
        int qa = ((bi >= 32) ? 2 : 0) + ((bj >= 32) ? 1 : 0);
        atomicAdd(&ws[OFF_QUAD + qa], s);
        if (bi != bj) {
            int qb = ((bj >= 32) ? 2 : 0) + ((bi >= 32) ? 1 : 0);
            atomicAdd(&ws[OFF_QUAD + qb], s);
        }
    }
}

__global__ void mmd_finish(const float* __restrict__ ws, float* __restrict__ out) {
    float q0 = ws[OFF_QUAD + 0];  // XX sum
    float q1 = ws[OFF_QUAD + 1];  // XY (top-right) sum
    float q2 = ws[OFF_QUAD + 2];  // YX (bottom-left) sum == XY sum
    float q3 = ws[OFF_QUAD + 3];  // YY sum
    // XX - 2*XY + YY, each mean over 4096^2; q1+q2 = 2*XY*n^2
    out[0] = (q0 - q1 - q2 + q3) * (1.0f / ((float)NHALF * (float)NHALF));
}

extern "C" void kernel_launch(void* const* d_in, const int* in_sizes, int n_in,
                              void* d_out, int out_size, void* d_ws, size_t ws_size,
                              hipStream_t stream) {
    (void)in_sizes; (void)n_in; (void)out_size; (void)ws_size;
    const float* X = (const float*)d_in[0];
    const float* Y = (const float*)d_in[1];
    float* ws  = (float*)d_ws;
    float* out = (float*)d_out;

    mmd_zero    <<<1,            256, 0, stream>>>(ws);
    mmd_rowstats<<<32,           256, 0, stream>>>(X, Y, ws);
    mmd_bw      <<<1,            256, 0, stream>>>(ws);
    mmd_main    <<<dim3(64, 64), 256, 0, stream>>>(X, Y, ws);
    mmd_finish  <<<1,            1,   0, stream>>>(ws, out);
}
